// Pointnet2StructurePointNet_15736760172886
// MI455X (gfx1250) — matrix operand order repaired and checked by reading, not compiled, not run on hardware
//
#include <hip/hip_runtime.h>
#include <hip/hip_bf16.h>

typedef __attribute__((ext_vector_type(16))) _Float16 v16h;
typedef __attribute__((ext_vector_type(8)))  _Float16 v8h;
typedef __attribute__((ext_vector_type(8)))  float    v8f;

// ---------------------------------------------------------------------------
// Farthest point sampling: one block per batch, block-wide argmax reduction.
// ---------------------------------------------------------------------------
__global__ void fps_kernel(const float* __restrict__ xyz, int n, int pstride,
                           int npoint, int* __restrict__ out) {
  int b = blockIdx.x;
  const float* base = xyz + (size_t)b * n * pstride;
  __shared__ float sd[256];
  __shared__ int   si[256];
  __shared__ int   s_last;
  int tid = threadIdx.x;
  float ld[32];
  int cnt = 0;
  for (int i = tid; i < n; i += 256) ld[cnt++] = 1e10f;
  if (tid == 0) { out[(size_t)b * npoint] = 0; s_last = 0; }
  __syncthreads();
  for (int it = 1; it < npoint; ++it) {
    int last = s_last;
    float px = base[(size_t)last * pstride + 0];
    float py = base[(size_t)last * pstride + 1];
    float pz = base[(size_t)last * pstride + 2];
    float best = -1.0f; int bidx = 0x7fffffff;
    cnt = 0;
    for (int i = tid; i < n; i += 256) {
      float dx = base[(size_t)i * pstride + 0] - px;
      float dy = base[(size_t)i * pstride + 1] - py;
      float dz = base[(size_t)i * pstride + 2] - pz;
      float d2 = dx * dx + dy * dy + dz * dz;
      float nd = fminf(ld[cnt], d2);
      ld[cnt] = nd; cnt++;
      if (nd > best) { best = nd; bidx = i; }
    }
    sd[tid] = best; si[tid] = bidx;
    __syncthreads();
    for (int o = 128; o > 0; o >>= 1) {
      if (tid < o) {
        float od = sd[tid + o]; int oi = si[tid + o];
        if (od > sd[tid] || (od == sd[tid] && oi < si[tid])) { sd[tid] = od; si[tid] = oi; }
      }
      __syncthreads();
    }
    if (tid == 0) { out[(size_t)b * npoint + it] = si[0]; s_last = si[0]; }
    __syncthreads();
  }
}

// ---------------------------------------------------------------------------
__global__ void gather_xyz_kernel(const float* __restrict__ xyz, int n, int pstride,
                                  const int* __restrict__ cidx, int npoint,
                                  float* __restrict__ out, int total) {
  int t = blockIdx.x * blockDim.x + threadIdx.x;
  if (t >= total) return;
  int d = t % 3, rest = t / 3;
  int s = rest % npoint, b = rest / npoint;
  int j = cidx[(size_t)b * npoint + s];
  out[t] = xyz[((size_t)b * n + j) * pstride + d];
}

// ---------------------------------------------------------------------------
// Ball query: one wave32 per center; ballot-compacted first-nsample-in-radius.
// ---------------------------------------------------------------------------
__global__ void ball_query_kernel(const float* __restrict__ xyz, int n, int pstride,
                                  const float* __restrict__ centers, int S,
                                  float r2, int ns, int* __restrict__ gidx, int nB) {
  int gt = blockIdx.x * blockDim.x + threadIdx.x;
  int wid = gt >> 5, lane = gt & 31;
  if (wid >= nB * S) return;
  int b = wid / S;
  const float* c = centers + (size_t)wid * 3;
  float cx = c[0], cy = c[1], cz = c[2];
  const float* base = xyz + (size_t)b * n * pstride;
  int* g = gidx + (size_t)wid * ns;
  int cnt = 0, first = -1;
  for (int j0 = 0; j0 < n && cnt < ns; j0 += 32) {
    int j = j0 + lane;
    bool within = false;
    if (j < n) {
      float dx = base[(size_t)j * pstride + 0] - cx;
      float dy = base[(size_t)j * pstride + 1] - cy;
      float dz = base[(size_t)j * pstride + 2] - cz;
      within = (dx * dx + dy * dy + dz * dz) < r2;
    }
    unsigned m = (unsigned)__ballot(within);
    if (first < 0 && m != 0u) first = j0 + __builtin_ctz(m);
    if (within) {
      unsigned lower = (lane == 0) ? 0u : (m & (0xffffffffu >> (32 - lane)));
      int pos = cnt + __builtin_popcount(lower);
      if (pos < ns) g[pos] = j;
    }
    cnt += __builtin_popcount(m);
  }
  if (first < 0) first = 0;
  if (cnt > ns) cnt = ns;
  for (int p = cnt + lane; p < ns; p += 32) g[p] = first;
}

// ---------------------------------------------------------------------------
// Build grouped f16 activations with zero-padded channel stride cinp.
// ---------------------------------------------------------------------------
__global__ void group_kernel(const float* __restrict__ xyz, int n, int pstride,
                             const float* __restrict__ newxyz, int S,
                             const float* __restrict__ feats, int fstride,
                             const int* __restrict__ gidx, int ns,
                             _Float16* __restrict__ X, int cin, int cinp,
                             int b, int total) {
  int t = blockIdx.x * blockDim.x + threadIdx.x;
  if (t >= total) return;
  int c = t % cinp, row = t / cinp;
  float v = 0.0f;
  if (c < cin) {
    int kk = row % ns, s = row / ns;
    int j = gidx[((size_t)b * S + s) * ns + kk];
    if (c < 3) v = xyz[((size_t)b * n + j) * pstride + c] - newxyz[((size_t)b * S + s) * 3 + c];
    else       v = feats[((size_t)b * n + j) * fstride + (c - 3)];
  }
  X[(size_t)row * cinp + c] = (_Float16)v;
}

// ---------------------------------------------------------------------------
// Weight f32 (O, cin) -> f16 (O, cinp) with zero pad.
// ---------------------------------------------------------------------------
__global__ void convert_w_pad_kernel(const float* __restrict__ W,
                                     _Float16* __restrict__ dst,
                                     int cin, int cinp, int total) {
  int t = blockIdx.x * blockDim.x + threadIdx.x;
  if (t >= total) return;
  int c = t % cinp, o = t / cinp;
  dst[t] = (c < cin) ? (_Float16)W[(size_t)o * cin + c] : (_Float16)0.0f;
}

// ---------------------------------------------------------------------------
__global__ void f32_to_f16_kernel(const float* __restrict__ src,
                                  _Float16* __restrict__ dst, int total) {
  int t = blockIdx.x * blockDim.x + threadIdx.x;
  if (t >= total) return;
  dst[t] = (_Float16)src[t];
}

// ---------------------------------------------------------------------------
// Zero pad columns [O, ldy) of an (M, ldy) f16 buffer.
// ---------------------------------------------------------------------------
__global__ void zero_cols_kernel(_Float16* __restrict__ Y, int M, int O, int ldy,
                                 int total) {
  int t = blockIdx.x * blockDim.x + threadIdx.x;
  if (t >= total) return;
  int w = ldy - O;
  int c = t % w, m = t / w;
  Y[(size_t)m * ldy + O + c] = (_Float16)0.0f;
}

// ---------------------------------------------------------------------------
// WMMA GEMM: Y[m,o] = gamma[o]*(sum_c X[m,c]*W[o,c] + bias[o]) + beta[o], ReLU opt.
// ldx is a multiple of 32 with zero padding -> unguarded b128 fragment loads.
// Each wave computes a 16x32 tile: one A fragment, two B fragments, two WMMAs.
// A frag (lane L, h=L>>4, m=L&15): halves K = h*8+0..7 and 16+h*8+0..7.
// B frag (lane L, n=L&15):         halves K = h*8+0..7 and 16+h*8+0..7 of row o0+n.
// ---------------------------------------------------------------------------
__global__ void gemm_wmma_kernel(const _Float16* __restrict__ X, int ldx,
                                 const _Float16* __restrict__ W,
                                 _Float16* __restrict__ Yh,
                                 float* __restrict__ Yf, int ldy,
                                 const float* __restrict__ gamma,
                                 const float* __restrict__ beta,
                                 const float* __restrict__ bias,
                                 int M, int O, int relu) {
  int wave = threadIdx.x >> 5;
  int lane = threadIdx.x & 31;
  int m0 = (blockIdx.x * 4 + wave) * 16;
  if (m0 >= M) return;                       // wave-uniform
  int o0 = blockIdx.y * 32;
  int h  = lane >> 4;
  int mn = lane & 15;
  int rowc = m0 + mn;      if (rowc > M - 1) rowc = M - 1;
  int w0   = o0 + mn;      if (w0   > O - 1) w0   = O - 1;
  int w1   = o0 + 16 + mn; if (w1   > O - 1) w1   = O - 1;
  const _Float16* xr  = X + (size_t)rowc * ldx + h * 8;
  const _Float16* wr0 = W + (size_t)w0 * ldx + h * 8;
  const _Float16* wr1 = W + (size_t)w1 * ldx + h * 8;
  v8f acc0 = {}, acc1 = {};
  for (int k0 = 0; k0 < ldx; k0 += 32) {
    union { v16h v; v8h q[2]; } a, b0, b1;
    a.q[0]  = *(const v8h*)(xr + k0);
    a.q[1]  = *(const v8h*)(xr + k0 + 16);
    b0.q[0] = *(const v8h*)(wr0 + k0);
    b0.q[1] = *(const v8h*)(wr0 + k0 + 16);
    b1.q[0] = *(const v8h*)(wr1 + k0);
    b1.q[1] = *(const v8h*)(wr1 + k0 + 16);
    acc0 = __builtin_amdgcn_wmma_f32_16x16x32_f16(false, a.v, false, b0.v,
                                                  (short)0, acc0, false, false);
    acc1 = __builtin_amdgcn_wmma_f32_16x16x32_f16(false, a.v, false, b1.v,
                                                  (short)0, acc1, false, false);
  }
#pragma unroll
  for (int half = 0; half < 2; ++half) {
    int ocol = o0 + half * 16 + mn;
    if (ocol >= O) continue;
    const v8f& acc = half ? acc1 : acc0;
    float gg = gamma[ocol];
    float bb = beta[ocol];
    float bs = bias ? bias[ocol] : 0.0f;
#pragma unroll
    for (int v = 0; v < 8; ++v) {
      int m = m0 + v + (h << 3);
      if (m >= M) continue;
      float y = gg * (acc[v] + bs) + bb;
      if (relu) y = fmaxf(y, 0.0f);
      if (Yh) Yh[(size_t)m * ldy + ocol] = (_Float16)y;
      if (Yf) Yf[(size_t)m * ldy + ocol] = y;
    }
  }
}

// ---------------------------------------------------------------------------
__global__ void maxpool_kernel(const _Float16* __restrict__ Xin, int S, int ns,
                               int cout, float* __restrict__ fout, int totC,
                               int choff, int b, int total) {
  int t = blockIdx.x * blockDim.x + threadIdx.x;
  if (t >= total) return;
  int c = t % cout, s = t / cout;
  const _Float16* p = Xin + (size_t)s * ns * cout + c;
  float m = -3.0e38f;
  for (int k = 0; k < ns; ++k) {
    float v = (float)p[(size_t)k * cout];
    m = fmaxf(m, v);
  }
  fout[((size_t)b * S + s) * totC + choff + c] = m;
}

// ---------------------------------------------------------------------------
__global__ void softmax_kernel(float* __restrict__ f, int S, int O, int nB) {
  int t = blockIdx.x * blockDim.x + threadIdx.x;
  if (t >= nB * O) return;
  int o = t % O, b = t / O;
  float* base = f + (size_t)b * S * O + o;
  float mx = -3.0e38f;
  for (int s = 0; s < S; ++s) mx = fmaxf(mx, base[(size_t)s * O]);
  float sum = 0.0f;
  for (int s = 0; s < S; ++s) sum += expf(base[(size_t)s * O] - mx);
  float inv = 1.0f / sum;
  for (int s = 0; s < S; ++s) base[(size_t)s * O] = expf(base[(size_t)s * O] - mx) * inv;
}

// ---------------------------------------------------------------------------
__global__ void final_kernel(const float* __restrict__ f,
                             const float* __restrict__ xyz2,
                             float* __restrict__ out, int S, int Mh, int nB) {
  int t = blockIdx.x * blockDim.x + threadIdx.x;
  if (t >= nB * Mh * 3) return;
  int d = t % 3, rest = t / 3;
  int m = rest % Mh, b = rest / Mh;
  float acc = 0.0f;
  for (int s = 0; s < S; ++s)
    acc += f[((size_t)b * S + s) * Mh + m] * xyz2[((size_t)b * S + s) * 3 + d];
  out[t] = acc;
}

// ===========================================================================
extern "C" void kernel_launch(void* const* d_in, const int* in_sizes, int n_in,
                              void* d_out, int out_size, void* d_ws, size_t ws_size,
                              hipStream_t stream) {
  (void)in_sizes; (void)n_in; (void)out_size; (void)ws_size;
  const int Bn = 4, Npts = 8192;
  int p = 0;
  const float* pc = (const float*)d_in[p++];

  struct SALayer { const float *W, *g, *b; int cin, cout; };
  static const int sa1_dims[3][4] = {{6,32,32,64},{6,64,64,128},{6,64,96,128}};
  static const int sa2_dims[3][4] = {{323,64,64,128},{323,128,128,256},{323,128,128,256}};
  SALayer sa1[3][3], sa2[3][3];
  for (int r = 0; r < 3; ++r)
    for (int l = 0; l < 3; ++l) {
      sa1[r][l].cin = sa1_dims[r][l]; sa1[r][l].cout = sa1_dims[r][l + 1];
      sa1[r][l].W = (const float*)d_in[p++];
      sa1[r][l].g = (const float*)d_in[p++];
      sa1[r][l].b = (const float*)d_in[p++];
    }
  for (int r = 0; r < 3; ++r)
    for (int l = 0; l < 3; ++l) {
      sa2[r][l].cin = sa2_dims[r][l]; sa2[r][l].cout = sa2_dims[r][l + 1];
      sa2[r][l].W = (const float*)d_in[p++];
      sa2[r][l].g = (const float*)d_in[p++];
      sa2[r][l].b = (const float*)d_in[p++];
    }
  struct HLayer { const float *W, *bias, *g, *b; int cin, cout; };
  static const int head_dims[8] = {640,512,256,128,64,32,16,16};
  HLayer head[7];
  for (int l = 0; l < 7; ++l) {
    head[l].cin = head_dims[l]; head[l].cout = head_dims[l + 1];
    head[l].W    = (const float*)d_in[p++];
    head[l].bias = (const float*)d_in[p++];
    head[l].g    = (const float*)d_in[p++];
    head[l].b    = (const float*)d_in[p++];
  }

  auto pad32 = [](int c) { return (c + 31) & ~31; };

  // ---- carve workspace ----
  char* wsb = (char*)d_ws;
  size_t off = 0;
  auto carve = [&](size_t bytes) -> void* {
    off = (off + 255) & ~(size_t)255;
    void* ptr = wsb + off;
    off += bytes;
    return ptr;
  };
  int*      cidx1  = (int*)carve((size_t)Bn * 512 * sizeof(int));
  float*    nx1    = (float*)carve((size_t)Bn * 512 * 3 * sizeof(float));
  int*      cidx2  = (int*)carve((size_t)Bn * 128 * sizeof(int));
  float*    nx2    = (float*)carve((size_t)Bn * 128 * 3 * sizeof(float));
  float*    feats1 = (float*)carve((size_t)Bn * 512 * 320 * sizeof(float));
  float*    feats2 = (float*)carve((size_t)Bn * 128 * 640 * sizeof(float));
  int*      gidx   = (int*)carve((size_t)Bn * 512 * 128 * sizeof(int));
  _Float16* wha    = (_Float16*)carve((size_t)1000000 * sizeof(_Float16));
  _Float16* Xa     = (_Float16*)carve((size_t)65536 * 128 * sizeof(_Float16));
  _Float16* Xb     = (_Float16*)carve((size_t)65536 * 128 * sizeof(_Float16));
  float*    Yf     = (float*)carve((size_t)512 * 16 * sizeof(float));

  size_t whoff = 0;
  auto wh_slot = [&](int count) { _Float16* s = wha + whoff; whoff += (size_t)count; return s; };
  auto blocks = [](int total) { return (total + 255) / 256; };

  // ---- one MSG set-abstraction stage ----
  auto run_sa = [&](const float* xyz, int n, int pstride,
                    const float* featsrc, int fstride, int featdim,
                    int npoint, const float* radii, const int* nss,
                    SALayer (*layers)[3], int* cidx, float* nxout,
                    float* featout, int totC) {
    fps_kernel<<<Bn, 256, 0, stream>>>(xyz, n, pstride, npoint, cidx);
    int tg = Bn * npoint * 3;
    gather_xyz_kernel<<<blocks(tg), 256, 0, stream>>>(xyz, n, pstride, cidx, npoint, nxout, tg);
    int choff = 0;
    for (int r = 0; r < 3; ++r) {
      int ns = nss[r];
      float r2 = radii[r] * radii[r];
      int waves = Bn * npoint;
      ball_query_kernel<<<blocks(waves * 32), 256, 0, stream>>>(
          xyz, n, pstride, nxout, npoint, r2, ns, gidx, Bn);
      _Float16* wh[3];
      for (int l = 0; l < 3; ++l) {
        SALayer& L = layers[r][l];
        int cinp = pad32(L.cin);
        int t2 = L.cout * cinp;
        wh[l] = wh_slot(t2);
        convert_w_pad_kernel<<<blocks(t2), 256, 0, stream>>>(L.W, wh[l], L.cin, cinp, t2);
      }
      int cin0  = 3 + featdim;
      int cin0p = pad32(cin0);
      for (int bi = 0; bi < Bn; ++bi) {
        int rows = npoint * ns;
        int tgr = rows * cin0p;
        group_kernel<<<blocks(tgr), 256, 0, stream>>>(
            xyz, n, pstride, nxout, npoint, featsrc, fstride, gidx, ns,
            Xa, cin0, cin0p, bi, tgr);
        _Float16* cur = Xa; _Float16* nxt = Xb;
        int ldx = cin0p;
        for (int l = 0; l < 3; ++l) {
          SALayer& L = layers[r][l];
          dim3 grid((rows + 63) / 64, (L.cout + 31) / 32);
          gemm_wmma_kernel<<<grid, 128, 0, stream>>>(
              cur, ldx, wh[l], nxt, nullptr, L.cout, L.g, L.b, nullptr,
              rows, L.cout, 1);
          _Float16* tsw = cur; cur = nxt; nxt = tsw;
          ldx = L.cout;                       // all SA couts are multiples of 32
        }
        int tm = npoint * ldx;
        maxpool_kernel<<<blocks(tm), 256, 0, stream>>>(
            cur, npoint, ns, ldx, featout, totC, choff, bi, tm);
      }
      choff += layers[r][2].cout;
    }
  };

  const float rad1[3] = {0.1f, 0.2f, 0.4f};
  const int   nsm1[3] = {16, 32, 128};
  const float rad2[3] = {0.2f, 0.4f, 0.8f};
  const int   nsm2[3] = {32, 64, 128};

  run_sa(pc, Npts, 6, pc + 3, 6, 3, 512, rad1, nsm1, sa1, cidx1, nx1, feats1, 320);
  run_sa(nx1, 512, 3, feats1, 320, 320, 128, rad2, nsm2, sa2, cidx2, nx2, feats2, 640);

  // ---- head: rows = b*128, layout (b*S, C) ----
  int trows = Bn * 128;
  int tf = trows * 640;
  f32_to_f16_kernel<<<blocks(tf), 256, 0, stream>>>(feats2, Xa, tf);
  _Float16* cur = Xa; _Float16* nxt = Xb;
  int ldx = 640;
  for (int l = 0; l < 7; ++l) {
    HLayer& L = head[l];
    int cinp = pad32(L.cin);
    int t2 = L.cout * cinp;
    _Float16* wh = wh_slot(t2);
    convert_w_pad_kernel<<<blocks(t2), 256, 0, stream>>>(L.W, wh, L.cin, cinp, t2);
    int is_last = (l == 6);
    int ldy = is_last ? L.cout : pad32(L.cout);
    dim3 grid((trows + 63) / 64, (L.cout + 31) / 32);
    gemm_wmma_kernel<<<grid, 128, 0, stream>>>(
        cur, ldx, wh, is_last ? nullptr : nxt, is_last ? Yf : nullptr, ldy,
        L.g, L.b, L.bias, trows, L.cout, is_last ? 0 : 1);
    if (!is_last) {
      if (ldy > L.cout) {
        int tz = trows * (ldy - L.cout);
        zero_cols_kernel<<<blocks(tz), 256, 0, stream>>>(nxt, trows, L.cout, ldy, tz);
      }
      _Float16* tsw = cur; cur = nxt; nxt = tsw;
      ldx = ldy;
    }
  }
  softmax_kernel<<<1, 64, 0, stream>>>(Yf, 128, 16, Bn);
  final_kernel<<<1, 192, 0, stream>>>(Yf, nx2, (float*)d_out, 128, 16, Bn);
}
